// L2AttentionLayer_28716151341359
// MI455X (gfx1250) — matrix operand order repaired and checked by reading, not compiled, not run on hardware
//
#include <hip/hip_runtime.h>
#include <hip/hip_bf16.h>

typedef __attribute__((ext_vector_type(16))) __bf16 v16bf;
typedef __attribute__((ext_vector_type(8)))  float  v8f;
typedef __attribute__((ext_vector_type(4)))  unsigned v4u;
typedef __attribute__((ext_vector_type(8)))  unsigned v8u;

constexpr int B  = 8;
constexpr int C  = 256;
constexpr int C4 = 64;
constexpr int N  = 2048;

__device__ __forceinline__ v8f wmma_bf16(v16bf a, v16bf b, v8f c) {
  // D = A(16x32 bf16) * B(32x16 bf16) + C(16x16 f32)
  return __builtin_amdgcn_wmma_f32_16x16x32_bf16(false, a, false, b, (short)0, c, false, false);
}

// A-fragment inverse maps (16x32 bf16 A operand, ISA 7.12.2):
__device__ __forceinline__ int a_e(int K) {
  return 2 * (((K >> 4) & 1) * 4 + ((K >> 1) & 3)) + (K & 1);
}
__device__ __forceinline__ int a_lh(int K) { return (K >> 3) & 1; }

__device__ __forceinline__ unsigned rfl(unsigned x) {
  return (unsigned)__builtin_amdgcn_readfirstlane((int)x);
}

// ---------------------------------------------------------------- converts
__global__ void convert_f32_bf16(const float* __restrict__ src,
                                 __bf16* __restrict__ dst, int n) {
  int i = blockIdx.x * blockDim.x + threadIdx.x;
  if (i < n) dst[i] = (__bf16)src[i];
}

// ------------------------------------------------- generic bf16 WMMA GEMM
__global__ __launch_bounds__(128)
void gemm_bf16_kernel(const __bf16* __restrict__ W, const __bf16* __restrict__ X,
                      const float* __restrict__ bias, float* __restrict__ outf,
                      __bf16* __restrict__ outb, int O, int K, int Nc) {
  __shared__ alignas(32) __bf16 wfrag[32][16];      // A-frag layout
  __shared__ alignas(32) __bf16 xfrag[4][32][16];   // B-frag layout per n-subtile
  const int tid = threadIdx.x, lane = tid & 31, wv = tid >> 5;
  const int half = lane >> 4, l15 = lane & 15;
  const int b  = blockIdx.z;
  const int o0 = blockIdx.y * 16;
  const int n0 = blockIdx.x * 64;
  const __bf16* Xb = X + (size_t)b * K * Nc;
  const unsigned short* Xs = (const unsigned short*)Xb;

  const int wM  = tid & 15;          // W-loader: o row (constant per thread)
  const int wK0 = (tid >> 4) * 2;    // W-loader: base K pair
  const int xn  = tid & 63;          // X-loader: n column (constant)
  const int xK0 = (tid >> 6) * 2;    // X-loader: base K pair

  v8f acc = {0.f, 0.f, 0.f, 0.f, 0.f, 0.f, 0.f, 0.f};
  for (int k0 = 0; k0 < K; k0 += 32) {
    __syncthreads();
#pragma unroll
    for (int k = 0; k < 2; ++k) {
      int Kp = wK0 + 16 * k;
      unsigned u = *(const unsigned*)(W + (size_t)(o0 + wM) * K + k0 + Kp);
      *(unsigned*)&wfrag[wM + 16 * a_lh(Kp)][a_e(Kp)] = u;
    }
#pragma unroll
    for (int k = 0; k < 8; ++k) {
      int Kp = xK0 + 4 * k;
      size_t gi = (size_t)(k0 + Kp) * Nc + n0 + xn;
      unsigned u = (unsigned)Xs[gi] | ((unsigned)Xs[gi + Nc] << 16);
      *(unsigned*)&xfrag[xn >> 4][(xn & 15) + 16 * a_lh(Kp)][a_e(Kp)] = u;
    }
    __syncthreads();
    v16bf af  = *(const v16bf*)&wfrag[lane][0];
    v16bf bfr = *(const v16bf*)&xfrag[wv][lane][0];
    acc = wmma_bf16(af, bfr, acc);
  }
#pragma unroll
  for (int r = 0; r < 8; ++r) {
    int o = o0 + r + 8 * half;
    int n = n0 + wv * 16 + l15;
    float val = acc[r] + (bias ? bias[o] : 0.f);
    size_t oi = ((size_t)b * O + o) * Nc + n;
    if (outf) outf[oi] = val;
    if (outb) outb[oi] = (__bf16)val;
  }
}

// ----------------------------------------------------------- ||q||^2 per point
__global__ void sq_kernel(const __bf16* __restrict__ q, float* __restrict__ sq) {
  int idx = blockIdx.x * blockDim.x + threadIdx.x;
  if (idx >= B * N) return;
  int b = idx / N, n = idx % N;
  float s = 0.f;
#pragma unroll 8
  for (int o = 0; o < C4; ++o) {
    float a = (float)q[((size_t)b * C4 + o) * N + n];
    s += a * a;
  }
  sq[idx] = s;
}

// ------------------------------- fused flash L2-attention: xr = softmax(-l2)*V
// Block = 128 thr (4 waves), 64 query rows; wave w owns 16 rows end-to-end.
// Score tile computed TRANSPOSED (A = q_j^T, B = q_i in regs) so the softmaxed
// P tile in registers is directly the A-fragment of O += P * V^T.
// V tile is brought in by the Tensor Data Mover: one tensor_load_to_lds per
// 32-j step (4D tile walk {32B, clo:16, jhalf:2, cgroup:16} lands linearly in
// the B-fragment LDS layout), waited with s_wait_tensorcnt.
__global__ __launch_bounds__(128)
void attn_kernel(const __bf16* __restrict__ q, const float* __restrict__ sq,
                 const __bf16* __restrict__ v, __bf16* __restrict__ xr) {
  __shared__ alignas(32) __bf16 qjfA[2][2][32][16]; // [kstep][jsub][lane][e] A-frag (rows=j)
  __shared__ alignas(32) __bf16 vjf[16][32][16];    // [ctile][lane][e]       B-frag (K=j,N=c)

  const int tid = threadIdx.x, lane = tid & 31, wv = tid >> 5;
  const int half = lane >> 4, l15 = lane & 15;
  const int b  = blockIdx.y;
  const int i0 = blockIdx.x * 64;
  const int ib = i0 + wv * 16;

  const __bf16* qg = q + (size_t)b * C4 * N;
  const unsigned short* qs = (const unsigned short*)qg;
  const __bf16* vg = v + (size_t)b * C * N;
  const float* sqg = sq + (size_t)b * N;

  // B operand q_i (32c x 16i per K-step): register-resident for whole kernel
  v16bf bqi0, bqi1;
#pragma unroll
  for (int e = 0; e < 16; ++e) {
    int c = (e & 7) + 8 * half + 16 * (e >> 3);
    bqi0[e] = qg[(size_t)c * N + ib + l15];
    bqi1[e] = qg[(size_t)(c + 32) * N + ib + l15];
  }
  float sqi = sqg[ib + l15];
  float mrow = -1e30f, lrow = 0.f;

  v8f oacc[16];
#pragma unroll
  for (int ct = 0; ct < 16; ++ct)
#pragma unroll
    for (int r = 0; r < 8; ++r) oacc[ct][r] = 0.f;

  // ---- TDM descriptor (constant groups), data_size = 1 byte ----
  // tile walk: X=32B (one 16-elem j-half of a c-row), dim1 = clo (16, stride
  // 4096B), dim2 = jhalf (2, stride 32B), dim3 = cgroup (16, stride 64KB).
  // LDS fill is exactly linear in B-frag order: 32B*16 = 512B per half,
  // 1KB per cgroup -> no padding required.
  const unsigned ldsv = rfl((unsigned)(size_t)&vjf[0][0][0]);
  const unsigned long long gvb = (unsigned long long)(size_t)vg;
  v8u g1;
  g1[0] = 0u;                     // wg_mask=0, data_size=0 (1B)
  g1[1] = (32u << 16);            // tensor_dim0 = 32 (bits 79:48, low half)
  g1[2] = (16u << 16);            // tensor_dim1 = 16 (bits 111:80, low half)
  g1[3] = (32u << 16);            // tile_dim0 = 32 (bits 127:112)
  g1[4] = 16u | (2u << 16);       // tile_dim1 = 16, tile_dim2 = 2
  g1[5] = 4096u;                  // tensor_dim0_stride = 4096 B
  g1[6] = (32u << 16);            // tensor_dim1_stride = 32 B (bits 255:208 low)
  g1[7] = 0u;
  v4u g2;
  g2[0] = 2u;                     // tensor_dim2 = 2
  g2[1] = 16u;                    // tensor_dim3 = 16
  g2[2] = 65536u;                 // tensor_dim2_stride = 64 KB
  g2[3] = (16u << 16);            // tile_dim3 = 16
  v4u g3;
  g3[0] = 1048576u;               // tensor_dim3_stride = 1 MB
  g3[1] = (1u << 16);             // tensor_dim4 = 1
  g3[2] = 0u;
  g3[3] = 0u;

  // manual-loader constants (q tile)
  const int qj  = tid & 31;          // j column
  const int qc0 = (tid >> 5) * 2;    // base even c (0,2,4,6)

  for (int jb = 0; jb < N; jb += 32) {
    __syncthreads();
    // V tile via Tensor Data Mover (wave 0 issues; TENSORcnt tracks)
    if (wv == 0) {
      unsigned long long ga = gvb + (unsigned long long)(jb * 2);
      v4u g0;
      g0[0] = 1u;                                           // count=1
      g0[1] = ldsv;                                         // lds_addr
      g0[2] = rfl((unsigned)ga);                            // global_addr lo
      g0[3] = rfl(((unsigned)(ga >> 32) & 0x01FFFFFFu) | 0x80000000u); // hi|type=2
      asm volatile("tensor_load_to_lds %0, %1, %2, %3"
                   :: "s"(g0), "s"(g1), "s"(g2), "s"(g3)
                   : "memory");
    }
    // q_j (64c x 32j) -> A-frags: pair along c (two strided b16 loads).
#pragma unroll
    for (int k = 0; k < 8; ++k) {
      int cp = qc0 + 8 * k;
      size_t gi = (size_t)cp * N + jb + qj;
      unsigned u = (unsigned)qs[gi] | ((unsigned)qs[gi + N] << 16);
      int cs = cp & 31;
      *(unsigned*)&qjfA[cp >> 5][(qj >> 3) & 1][(qj & 7) + 8 * (qj >> 4) + 16 * a_lh(cs)][a_e(cs)] = u;
    }
    if (jb + 32 < N)
      __builtin_prefetch(vg + (size_t)(tid * 2) * N + jb + 32, 0, 1);
    if (wv == 0) __builtin_amdgcn_s_wait_tensorcnt(0);
    __syncthreads();

    // S^T tiles: D rows = j (s0: r + 16*half, s1: 8 + r + 16*half), cols = i (l15)
    v8f s0 = {0.f,0.f,0.f,0.f,0.f,0.f,0.f,0.f};
    v8f s1 = {0.f,0.f,0.f,0.f,0.f,0.f,0.f,0.f};
    {
      v16bf a;
      a = *(const v16bf*)&qjfA[0][0][lane][0]; s0 = wmma_bf16(a, bqi0, s0);
      a = *(const v16bf*)&qjfA[0][1][lane][0]; s1 = wmma_bf16(a, bqi0, s1);
      a = *(const v16bf*)&qjfA[1][0][lane][0]; s0 = wmma_bf16(a, bqi1, s0);
      a = *(const v16bf*)&qjfA[1][1][lane][0]; s1 = wmma_bf16(a, bqi1, s1);
    }

    float sqjv = sqg[jb + lane];     // lane <-> j within block
    float t0[8], t1[8];
    float mx = -1e30f;
#pragma unroll
    for (int r = 0; r < 8; ++r) {
      int j0 = r + 16 * half;
      float sj0 = __shfl(sqjv, j0, 32);
      float sj1 = __shfl(sqjv, 8 + j0, 32);
      t0[r] = -sqrtf(fmaxf(sqi + sj0 - 2.f * s0[r], 0.f));
      t1[r] = -sqrtf(fmaxf(sqi + sj1 - 2.f * s1[r], 0.f));
      mx = fmaxf(mx, fmaxf(t0[r], t1[r]));
    }
    mx = fmaxf(mx, __shfl_xor(mx, 16, 32));   // join both halves of column i
    float mnew = fmaxf(mrow, mx);
    float sc = __expf(mrow - mnew);
    bool changed = (mnew != mrow);
    float rs = 0.f;
    v16bf pf;                                  // P directly in A-frag layout
#pragma unroll
    for (int r = 0; r < 8; ++r) {
      float p0 = __expf(t0[r] - mnew);
      float p1 = __expf(t1[r] - mnew);
      rs += p0 + p1;
      pf[r]     = (__bf16)p0;   // e=r   <-> j = r + 16*half
      pf[r + 8] = (__bf16)p1;   // e=r+8 <-> j = 8 + r + 16*half
    }
    rs += __shfl_xor(rs, 16, 32);
    lrow = lrow * sc + rs;
    mrow = mnew;

    // rescale only when the running max actually moved (uniform branch)
    if (__ballot(changed) != 0ull) {
      float scr[8];
#pragma unroll
      for (int r = 0; r < 8; ++r) scr[r] = __shfl(sc, r + 8 * half, 32);
#pragma unroll
      for (int ct = 0; ct < 16; ++ct)
#pragma unroll
        for (int r = 0; r < 8; ++r) oacc[ct][r] *= scr[r];
    }

#pragma unroll
    for (int ct = 0; ct < 16; ++ct) {
      v16bf vb = *(const v16bf*)&vjf[ct][lane][0];
      oacc[ct] = wmma_bf16(pf, vb, oacc[ct]);
    }
  }

  float linvr[8];
#pragma unroll
  for (int r = 0; r < 8; ++r)
    linvr[r] = 1.f / __shfl(lrow, r + 8 * half, 32);

#pragma unroll
  for (int ct = 0; ct < 16; ++ct) {
    int c = ct * 16 + l15;
#pragma unroll
    for (int r = 0; r < 8; ++r) {
      int i = ib + r + 8 * half;
      xr[((size_t)b * C + c) * N + i] = (__bf16)(oacc[ct][r] * linvr[r]);
    }
  }
}

// ------------------------------------------------------------- BN statistics
__global__ __launch_bounds__(256)
void bnstats_kernel(const float* __restrict__ t, float* __restrict__ mean,
                    float* __restrict__ rstd) {
  __shared__ float rs[256], rs2[256];
  const int c = blockIdx.x, tid = threadIdx.x;
  float s = 0.f, s2 = 0.f;
  for (int i = tid; i < B * N; i += 256) {
    int b = i / N, n = i % N;
    float v = t[((size_t)b * C + c) * N + n];
    s += v; s2 += v * v;
  }
  rs[tid] = s; rs2[tid] = s2;
  __syncthreads();
  for (int off = 128; off > 0; off >>= 1) {
    if (tid < off) { rs[tid] += rs[tid + off]; rs2[tid] += rs2[tid + off]; }
    __syncthreads();
  }
  if (tid == 0) {
    float inv = 1.f / (float)(B * N);
    float m = rs[0] * inv;
    float var = rs2[0] * inv - m * m;
    mean[c] = m;
    rstd[c] = rsqrtf(var + 1e-5f);
  }
}

// ------------------------------------------- BN apply + ReLU + residual add
__global__ void final_kernel(const float* __restrict__ x, const float* __restrict__ t,
                             const float* __restrict__ mean, const float* __restrict__ rstd,
                             const float* __restrict__ gamma, const float* __restrict__ beta,
                             float* __restrict__ out) {
  int idx = blockIdx.x * blockDim.x + threadIdx.x;
  if (idx >= B * C * N) return;
  int c = (idx / N) % C;
  float bn = gamma[c] * (t[idx] - mean[c]) * rstd[c] + beta[c];
  out[idx] = x[idx] + fmaxf(bn, 0.f);
}

// ---------------------------------------------------------------------------
extern "C" void kernel_launch(void* const* d_in, const int* in_sizes, int n_in,
                              void* d_out, int out_size, void* d_ws, size_t ws_size,
                              hipStream_t stream) {
  const float* x     = (const float*)d_in[0];
  const float* wq    = (const float*)d_in[1];
  const float* wvw   = (const float*)d_in[2];
  const float* bv    = (const float*)d_in[3];
  const float* wtw   = (const float*)d_in[4];
  const float* bt    = (const float*)d_in[5];
  const float* gamma = (const float*)d_in[6];
  const float* beta  = (const float*)d_in[7];
  float* out = (float*)d_out;

  char* wsp = (char*)d_ws;
  auto alloc = [&](size_t bytes) -> char* {
    char* p = wsp;
    wsp += (bytes + 255) & ~(size_t)255;
    return p;
  };
  __bf16* xbf  = (__bf16*)alloc((size_t)B * C * N * 2);
  __bf16* wqb  = (__bf16*)alloc((size_t)C4 * C * 2);
  __bf16* wvb  = (__bf16*)alloc((size_t)C * C * 2);
  __bf16* wtb  = (__bf16*)alloc((size_t)C * C * 2);
  __bf16* qbf  = (__bf16*)alloc((size_t)B * C4 * N * 2);
  float*  sqb  = (float*) alloc((size_t)B * N * 4);
  __bf16* vbf  = (__bf16*)alloc((size_t)B * C * N * 2);
  __bf16* xrbf = (__bf16*)alloc((size_t)B * C * N * 2);
  float*  tbuf = (float*) alloc((size_t)B * C * N * 4);
  float*  mean = (float*) alloc(C * 4);
  float*  rstd = (float*) alloc(C * 4);

  convert_f32_bf16<<<(B * C * N) / 256, 256, 0, stream>>>(x, xbf, B * C * N);
  convert_f32_bf16<<<(C4 * C) / 256, 256, 0, stream>>>(wq, wqb, C4 * C);
  convert_f32_bf16<<<(C * C) / 256, 256, 0, stream>>>(wvw, wvb, C * C);
  convert_f32_bf16<<<(C * C) / 256, 256, 0, stream>>>(wtw, wtb, C * C);

  gemm_bf16_kernel<<<dim3(N / 64, C4 / 16, B), 128, 0, stream>>>(
      wqb, xbf, nullptr, nullptr, qbf, C4, C, N);
  gemm_bf16_kernel<<<dim3(N / 64, C / 16, B), 128, 0, stream>>>(
      wvb, xbf, bv, nullptr, vbf, C, C, N);

  sq_kernel<<<(B * N) / 256, 256, 0, stream>>>(qbf, sqb);

  attn_kernel<<<dim3(N / 64, B), 128, 0, stream>>>(qbf, sqb, vbf, xrbf);

  gemm_bf16_kernel<<<dim3(N / 64, C / 16, B), 128, 0, stream>>>(
      wtb, xrbf, bt, tbuf, nullptr, C, C, N);

  bnstats_kernel<<<C, 256, 0, stream>>>(tbuf, mean, rstd);
  final_kernel<<<(B * C * N) / 256, 256, 0, stream>>>(
      x, tbuf, mean, rstd, gamma, beta, out);

  (void)in_sizes; (void)n_in; (void)out_size; (void)ws_size;
}
